// CNNGNNModel_32530082299962
// MI455X (gfx1250) — matrix operand order repaired and checked by reading, not compiled, not run on hardware
//
#include <hip/hip_runtime.h>
#include <hip/hip_bf16.h>

// ---------- types ----------
typedef _Float16 v8h  __attribute__((ext_vector_type(8)));
typedef _Float16 v16h __attribute__((ext_vector_type(16)));
typedef float    v8f  __attribute__((ext_vector_type(8)));

#define BATCH   64
#define LIN     16384
#define NPG     4096          // nodes per graph (LIN/4)
#define NTOT    (BATCH * NPG) // 262144
#define QLEN    8192          // after first pool

// =====================================================================
// Kernel 0: pack GCN weights (f32 row-major [K,64]) into WMMA B-fragment
// lane order, converted to f16.  W1 frags (K padded 8->32): ids 0..3.
// W2 frags (K=64, 2 k-blocks): ids 0..7 at offset 2048.
// B-frag packing used here (contiguous k, A matches): lane<16 -> col=lane, K=kbase+0..15 ;
// lane>=16 -> col=lane-16, K=kbase+16..31 ; element e = K kbase+e(+16).
// =====================================================================
__global__ void cg_pack_w(const float* __restrict__ w1,
                          const float* __restrict__ w2,
                          _Float16* __restrict__ wpack) {
    int tid = blockIdx.x * blockDim.x + threadIdx.x;   // 0..6143
    if (tid < 2048) {                                  // W1: 4 frags * 512
        int nt = tid >> 9, r = tid & 511;
        int lane = r >> 4, e = r & 15;
        int col = nt * 16 + (lane & 15);
        int k = ((lane < 16) ? 0 : 16) + e;
        float v = (k < 8) ? w1[k * 64 + col] : 0.0f;
        wpack[tid] = (_Float16)v;
    } else if (tid < 6144) {                           // W2: 8 frags * 512
        int id = (tid - 2048) >> 9, r = tid & 511;
        int kk = id >> 2, nt = id & 3;
        int lane = r >> 4, e = r & 15;
        int col = nt * 16 + (lane & 15);
        int k = kk * 32 + ((lane < 16) ? 0 : 16) + e;
        wpack[tid] = (_Float16)w2[k * 64 + col];
    }
}

// =====================================================================
// Kernel 1: fused conv1+relu+pool2 + conv2+relu+pool2 -> nodes [N,8] f16
// One block = 256 output nodes of one graph. x tile staged in LDS,
// stage-1 result staged in LDS, never spilled to HBM.
// =====================================================================
__global__ void cg_conv_fused(const float* __restrict__ x,
                              const float* __restrict__ w1,
                              const float* __restrict__ b1,
                              const float* __restrict__ w2,
                              const float* __restrict__ b2,
                              _Float16* __restrict__ nodes) {
    __shared__ float xs[1032];
    __shared__ float ys[8][516];
    __shared__ float lw1[24], lb1[8], lw2[192], lb2[8];
    const int t  = threadIdx.x;
    const int b  = blockIdx.x >> 4;          // graph
    const int p0 = (blockIdx.x & 15) * 256;  // first node of tile
    if (t < 24)  lw1[t] = w1[t];
    if (t < 8)   { lb1[t] = b1[t]; lb2[t] = b2[t]; }
    if (t < 192) lw2[t] = w2[t];
    // x tile with halo (conv pad handled by zeros)
    const int xbase = 4 * p0 - 3;
    const float* xb = x + (size_t)b * LIN;
    for (int g = t; g < 1030; g += 256) {
        int tt = xbase + g;
        xs[g] = (tt >= 0 && tt < LIN) ? xb[tt] : 0.0f;
    }
    __syncthreads();
    // stage 1: y1[q] = relu(max(conv1@2q, conv1@2q+1) + b1), q in tile+halo
    for (int e = t; e < 8 * 514; e += 256) {
        int qi = e >> 3, c = e & 7;
        int q = 2 * p0 - 1 + qi;
        float v = 0.0f;
        if (q >= 0 && q < QLEN) {
            int ix = 2 * qi;
            float w0 = lw1[c*3], wA = lw1[c*3+1], wB = lw1[c*3+2];
            float va = w0*xs[ix]   + wA*xs[ix+1] + wB*xs[ix+2];
            float vb = w0*xs[ix+1] + wA*xs[ix+2] + wB*xs[ix+3];
            v = fmaxf(fmaxf(va, vb) + lb1[c], 0.0f);
        }
        ys[c][qi] = v;
    }
    __syncthreads();
    // stage 2: one node per thread, 8 output channels
    float yv[8][4];
#pragma unroll
    for (int c1 = 0; c1 < 8; ++c1)
#pragma unroll
        for (int o = 0; o < 4; ++o) yv[c1][o] = ys[c1][2 * t + o];
    v8h ov;
#pragma unroll
    for (int c2 = 0; c2 < 8; ++c2) {
        float a0 = 0.0f, a1 = 0.0f;
#pragma unroll
        for (int c1 = 0; c1 < 8; ++c1) {
            const float w0 = lw2[(c2*8+c1)*3+0];
            const float wA = lw2[(c2*8+c1)*3+1];
            const float wB = lw2[(c2*8+c1)*3+2];
            a0 += w0*yv[c1][0] + wA*yv[c1][1] + wB*yv[c1][2];
            a1 += w0*yv[c1][1] + wA*yv[c1][2] + wB*yv[c1][3];
        }
        ov[c2] = (_Float16)fmaxf(fmaxf(a0, a1) + lb2[c2], 0.0f);
    }
    *(v8h*)(nodes + (size_t)(b * NPG + p0 + t) * 8) = ov;
}

// ---------- GCN normalization coefficients for chain graph + self loop ----
// agg[i] = cs*h[i] + cl*h[i-1] + cr*h[i+1]  (coeff 0 at graph endpoints)
__device__ __forceinline__ void gcn_coeffs_h(int p, _Float16& cs, _Float16& cl,
                                             _Float16& cr) {
    float d  = (p == 0 || p == NPG - 1) ? 2.0f : 3.0f;
    float fs = 1.0f / d;
    float fl = (p > 0)       ? __frsqrt_rn(((p == 1)       ? 2.0f : 3.0f) * d) : 0.0f;
    float fr = (p < NPG - 1) ? __frsqrt_rn(((p == NPG - 2) ? 2.0f : 3.0f) * d) : 0.0f;
    cs = (_Float16)fs; cl = (_Float16)fl; cr = (_Float16)fr;
}

__device__ __forceinline__ v8h stencil3(v8h zc, v8h zl, v8h zr,
                                        _Float16 cs, _Float16 cl, _Float16 cr) {
    v8h o;
#pragma unroll
    for (int e = 0; e < 8; ++e)
        o[e] = cs * zc[e] + cl * zl[e] + cr * zr[e];   // packed f16 fma
    return o;
}

// =====================================================================
// Kernel 2: GCN1 via WMMA with in-register stencil:
//   h1 = relu( stencil(nodes)[N,8] @ W1[8,64] + b1 )
// (stencil commutes with the matmul since it is linear over rows)
// K=8 zero-padded into one v_wmma_f32_16x16x32_f16 per 16-col tile.
// =====================================================================
__global__ void cg_gemm_gcn1(const _Float16* __restrict__ nodes,
                             const _Float16* __restrict__ wpack,
                             const float* __restrict__ bias,
                             _Float16* __restrict__ out) {
    const int wave = (blockIdx.x * blockDim.x + threadIdx.x) >> 5;  // M-tile
    const int lane = threadIdx.x & 31;
    const int m0   = wave * 16;
    const bool lo  = lane < 16;
    const int row  = m0 + (lane & 15);
    const int p    = row & (NPG - 1);
    const int im   = (p > 0)       ? row - 1 : row;  // clamped; coeff 0 anyway
    const int ip   = (p < NPG - 1) ? row + 1 : row;
    _Float16 cs, cl, cr; gcn_coeffs_h(p, cs, cl, cr);
    v8h zc = *(const v8h*)(nodes + (size_t)row * 8);
    v8h zl = *(const v8h*)(nodes + (size_t)im  * 8);
    v8h zr = *(const v8h*)(nodes + (size_t)ip  * 8);
    v8h ar = stencil3(zc, zl, zr, cs, cl, cr);
    v16h a = {};
#pragma unroll
    for (int e = 0; e < 8; ++e) a[e] = lo ? ar[e] : (_Float16)0.0f;
#pragma unroll
    for (int nt = 0; nt < 4; ++nt) {
        const v8h* bp = (const v8h*)(wpack + nt * 512 + lane * 16);
        v8h b0 = bp[0], b1v = bp[1];
        v16h bf;
#pragma unroll
        for (int e = 0; e < 8; ++e) { bf[e] = b0[e]; bf[8 + e] = b1v[e]; }
        v8f c = {};
        c = __builtin_amdgcn_wmma_f32_16x16x32_f16(false, a, false, bf,
                                                   (short)0, c, false, false);
        const int col = nt * 16 + (lane & 15);
        const float bv = bias[col];
        const int rbase = m0 + (lo ? 0 : 8);
#pragma unroll
        for (int r = 0; r < 8; ++r) {
            float v = fmaxf(c[r] + bv, 0.0f);
            out[(size_t)(rbase + r) * 64 + col] = (_Float16)v;
        }
    }
}

// =====================================================================
// Kernel 3: GCN2 via WMMA with in-register stencil:
//   h2 = relu( stencil(h1)[N,64] @ W2[64,64] + b2 )
// K=64 -> two accumulated 16x16x32 WMMAs per 16x16 output tile.
// Neighbor-row chunk loads hit L0/L2 (rows are shared across tiles).
// =====================================================================
__global__ void cg_gemm_gcn2(const _Float16* __restrict__ h1,
                             const _Float16* __restrict__ wpack2,
                             const float* __restrict__ bias,
                             _Float16* __restrict__ out) {
    const int wave = (blockIdx.x * blockDim.x + threadIdx.x) >> 5;
    const int lane = threadIdx.x & 31;
    const int m0   = wave * 16;
    const bool lo  = lane < 16;
    const int row  = m0 + (lane & 15);
    const int p    = row & (NPG - 1);
    const int im   = (p > 0)       ? row - 1 : row;
    const int ip   = (p < NPG - 1) ? row + 1 : row;
    _Float16 cs, cl, cr; gcn_coeffs_h(p, cs, cl, cr);
    const int koff = lo ? 0 : 16;
    const _Float16* rc = h1 + (size_t)row * 64;
    const _Float16* rl = h1 + (size_t)im  * 64;
    const _Float16* rr = h1 + (size_t)ip  * 64;
    // 4 chunks per row: k-block {0,32} x element-half {0,8}, at koff
    v8h s0, s1v, s2, s3;
    {
        v8h c0 = *(const v8h*)(rc + koff),        l0 = *(const v8h*)(rl + koff),
            r0 = *(const v8h*)(rr + koff);
        v8h c1 = *(const v8h*)(rc + koff + 8),   l1 = *(const v8h*)(rl + koff + 8),
            r1 = *(const v8h*)(rr + koff + 8);
        v8h c2 = *(const v8h*)(rc + 32 + koff),   l2 = *(const v8h*)(rl + 32 + koff),
            r2 = *(const v8h*)(rr + 32 + koff);
        v8h c3 = *(const v8h*)(rc + 40 + koff),   l3 = *(const v8h*)(rl + 40 + koff),
            r3 = *(const v8h*)(rr + 40 + koff);
        s0 = stencil3(c0, l0, r0, cs, cl, cr);
        s1v = stencil3(c1, l1, r1, cs, cl, cr);
        s2 = stencil3(c2, l2, r2, cs, cl, cr);
        s3 = stencil3(c3, l3, r3, cs, cl, cr);
    }
    v16h A0, A1;
#pragma unroll
    for (int e = 0; e < 8; ++e) {
        A0[e] = s0[e]; A0[8 + e] = s1v[e];
        A1[e] = s2[e]; A1[8 + e] = s3[e];
    }
#pragma unroll
    for (int nt = 0; nt < 4; ++nt) {
        const v8h* bp0 = (const v8h*)(wpack2 + (0 * 4 + nt) * 512 + lane * 16);
        const v8h* bp1 = (const v8h*)(wpack2 + (1 * 4 + nt) * 512 + lane * 16);
        v8h b00 = bp0[0], b01 = bp0[1], b10 = bp1[0], b11 = bp1[1];
        v16h B0, B1;
#pragma unroll
        for (int e = 0; e < 8; ++e) {
            B0[e] = b00[e]; B0[8 + e] = b01[e];
            B1[e] = b10[e]; B1[8 + e] = b11[e];
        }
        v8f c = {};
        c = __builtin_amdgcn_wmma_f32_16x16x32_f16(false, A0, false, B0,
                                                   (short)0, c, false, false);
        c = __builtin_amdgcn_wmma_f32_16x16x32_f16(false, A1, false, B1,
                                                   (short)0, c, false, false);
        const int col = nt * 16 + (lane & 15);
        const float bv = bias[col];
        const int rbase = m0 + (lo ? 0 : 8);
#pragma unroll
        for (int r = 0; r < 8; ++r) {
            float v = fmaxf(c[r] + bv, 0.0f);
            out[(size_t)(rbase + r) * 64 + col] = (_Float16)v;
        }
    }
}

// =====================================================================
// Kernel 4: mean-pool over 4096 nodes per graph + fc(64->2)
// One block per graph; thread t sums column t&63 over a quarter of rows.
// Deterministic reduction (no atomics).
// =====================================================================
__global__ void cg_pool_fc(const _Float16* __restrict__ h2,
                           const float* __restrict__ fcw,
                           const float* __restrict__ fcb,
                           float* __restrict__ out) {
    __shared__ float red[256];
    __shared__ float pooled[64];
    const int g = blockIdx.x, t = threadIdx.x;
    const int col = t & 63, part = t >> 6;
    const _Float16* base = h2 + ((size_t)g * NPG + part * 1024) * 64 + col;
    float s = 0.0f;
    for (int r = 0; r < 1024; ++r) s += (float)base[(size_t)r * 64];
    red[t] = s;
    __syncthreads();
    if (t < 64)
        pooled[t] = (red[t] + red[t + 64] + red[t + 128] + red[t + 192]) *
                    (1.0f / (float)NPG);
    __syncthreads();
    if (t < 2) {
        float acc = fcb[t];
        for (int k = 0; k < 64; ++k) acc += pooled[k] * fcw[k * 2 + t];
        out[g * 2 + t] = acc;
    }
}

// =====================================================================
extern "C" void kernel_launch(void* const* d_in, const int* in_sizes, int n_in,
                              void* d_out, int out_size, void* d_ws, size_t ws_size,
                              hipStream_t stream) {
    const float* x       = (const float*)d_in[0];
    const float* conv1_w = (const float*)d_in[2];
    const float* conv1_b = (const float*)d_in[3];
    const float* conv2_w = (const float*)d_in[4];
    const float* conv2_b = (const float*)d_in[5];
    const float* gcn1_w  = (const float*)d_in[6];
    const float* gcn1_b  = (const float*)d_in[7];
    const float* gcn2_w  = (const float*)d_in[8];
    const float* gcn2_b  = (const float*)d_in[9];
    const float* fc_w    = (const float*)d_in[10];
    const float* fc_b    = (const float*)d_in[11];
    float* out = (float*)d_out;

    // workspace layout (bytes)
    char* ws = (char*)d_ws;
    _Float16* nodes = (_Float16*)(ws + 0);                  //  4 MB  [N,8]
    _Float16* h1    = (_Float16*)(ws + (size_t)4194304);    // 32 MB  [N,64]
    _Float16* h2    = (_Float16*)(ws + (size_t)37748736);   // 32 MB  [N,64]
    _Float16* wpack = (_Float16*)(ws + (size_t)71303168);   // 12 KB packed B-frags

    cg_pack_w    <<<24,   256, 0, stream>>>(gcn1_w, gcn2_w, wpack);
    cg_conv_fused<<<1024, 256, 0, stream>>>(x, conv1_w, conv1_b,
                                            conv2_w, conv2_b, nodes);
    cg_gemm_gcn1 <<<NTOT / 16 / 8, 256, 0, stream>>>(nodes, wpack, gcn1_b, h1);
    cg_gemm_gcn2 <<<NTOT / 16 / 8, 256, 0, stream>>>(h1, wpack + 2048, gcn2_b, h2);
    cg_pool_fc   <<<BATCH,         256, 0, stream>>>(h2, fc_w, fc_b, out);
}
